// SelfAttentionWrapper_19275813224540
// MI455X (gfx1250) — matrix operand order repaired and checked by reading, not compiled, not run on hardware
//
#include <hip/hip_runtime.h>
#include <hip/hip_bf16.h>
#include <math.h>

// ---------------- types ----------------
typedef __attribute__((ext_vector_type(16))) __bf16 v16bf;
typedef __attribute__((ext_vector_type(8)))  __bf16 v8bf;
typedef __attribute__((ext_vector_type(8)))  float  v8f;

#define D_MODEL  2048
#define N_HEAD   16
#define HEAD_DIM 128
#define SEQ      2048
#define BATCH    2
#define M_TOK    (BATCH * SEQ)   // 4096 tokens

// ---------------- CDNA5 async copy (ASYNCcnt-tracked DMA to LDS) ----------------
// Each lane copies 16 bytes global -> LDS. lds_off is the byte offset within the
// workgroup LDS allocation (low 32 bits of a generic pointer to __shared__).
__device__ __forceinline__ void async_b128(unsigned int lds_off, const void* gaddr) {
  asm volatile("global_load_async_to_lds_b128 %0, %1, off"
               :: "v"(lds_off), "v"(gaddr) : "memory");
}
__device__ __forceinline__ void wait_async0() {
  asm volatile("s_wait_asynccnt 0" ::: "memory");
}

// ---------------- WMMA helpers ----------------
__device__ __forceinline__ v8f wmma_bf16(v16bf a, v16bf b, v8f c) {
  // emits v_wmma_f32_16x16x32_bf16
  return __builtin_amdgcn_wmma_f32_16x16x32_bf16(false, a, false, b, (short)0, c,
                                                 false, false);
}

// A-fragment (16x32 bf16, MxK) from row-major [rows][ld] source.
// CDNA5 layout: lanes 0-15 M=lane, K=0..7 & 16..23; lanes 16-31 K=8..15 & 24..31.
__device__ __forceinline__ v16bf load_afrag(const __bf16* base, int ld, int row0, int k0) {
  const int lane = threadIdx.x & 31;
  const int r    = row0 + (lane & 15);
  const int koff = (lane >> 4) << 3;           // 0 or 8
  const __bf16* p = base + (size_t)r * ld + k0 + koff;
  v8bf lo = *reinterpret_cast<const v8bf*>(p);
  v8bf hi = *reinterpret_cast<const v8bf*>(p + 16);
  v16bf out;
#pragma unroll
  for (int i = 0; i < 8; ++i) { out[i] = lo[i]; out[i + 8] = hi[i]; }
  return out;
}

// B-fragment (32x16 bf16, KxN) given the matrix stored as N-major [n][k] rows
// (i.e. B^T row-major). Lanes 0-15 hold K=0..7 & 16..23, lanes 16-31 hold K=8..15 & 24..31.
__device__ __forceinline__ v16bf load_bfrag(const __bf16* base, int ld, int row0, int k0) {
  const int lane = threadIdx.x & 31;
  const int r    = row0 + (lane & 15);
  const int kb   = (lane >> 4) << 3;           // 0 or 8
  const __bf16* p = base + (size_t)r * ld + k0 + kb;
  v8bf lo = *reinterpret_cast<const v8bf*>(p);
  v8bf hi = *reinterpret_cast<const v8bf*>(p + 16);
  v16bf out;
#pragma unroll
  for (int i = 0; i < 8; ++i) { out[i] = lo[i]; out[i + 8] = hi[i]; }
  return out;
}

// ---------------- 1) residual add + RMSNorm -> bf16 activations ----------------
__global__ __launch_bounds__(256) void add_rmsnorm_kernel(
    const float* __restrict__ h, const float* __restrict__ rin,
    const float* __restrict__ w, float* __restrict__ resid_out,
    __bf16* __restrict__ xb)
{
  const int row = blockIdx.x;            // token
  const int tid = threadIdx.x;
  const size_t base = (size_t)row * D_MODEL;
  float vals[8];
  float ss = 0.f;
#pragma unroll
  for (int i = 0; i < 8; ++i) {
    int c = tid + i * 256;
    float v = h[base + c] + rin[base + c];
    vals[i] = v;
    resid_out[base + c] = v;             // second tuple output (fp32)
    ss += v * v;
  }
#pragma unroll
  for (int off = 16; off > 0; off >>= 1) ss += __shfl_xor(ss, off, 32);
  __shared__ float wss[8];
  if ((tid & 31) == 0) wss[tid >> 5] = ss;
  __syncthreads();
  float tot = 0.f;
#pragma unroll
  for (int j = 0; j < 8; ++j) tot += wss[j];
  const float rms = rsqrtf(tot * (1.0f / D_MODEL) + 1e-5f);
#pragma unroll
  for (int i = 0; i < 8; ++i) {
    int c = tid + i * 256;
    xb[base + c] = (__bf16)(vals[i] * rms * w[c]);
  }
}

// ---------------- 2) weight transpose fp32 -> bf16, N-major ----------------
__global__ __launch_bounds__(256) void transpose_w_kernel(
    const float* __restrict__ W0, const float* __restrict__ W1,
    const float* __restrict__ W2, const float* __restrict__ W3,
    __bf16* __restrict__ T0, __bf16* __restrict__ T1,
    __bf16* __restrict__ T2, __bf16* __restrict__ T3)
{
  __shared__ float tile[32][33];
  const float* W; __bf16* T;
  switch (blockIdx.z) {
    case 0:  W = W0; T = T0; break;
    case 1:  W = W1; T = T1; break;
    case 2:  W = W2; T = T2; break;
    default: W = W3; T = T3; break;
  }
  const int k0 = blockIdx.y * 32, n0 = blockIdx.x * 32;
  const int tx = threadIdx.x, ty = threadIdx.y;     // (32, 8)
#pragma unroll
  for (int i = 0; i < 4; ++i)
    tile[ty + 8 * i][tx] = W[(size_t)(k0 + ty + 8 * i) * D_MODEL + n0 + tx];
  __syncthreads();
#pragma unroll
  for (int i = 0; i < 4; ++i)
    T[(size_t)(n0 + ty + 8 * i) * D_MODEL + k0 + tx] = (__bf16)tile[tx][ty + 8 * i];
}

// ---------------- 3) WMMA GEMM, LDS double-buffered via async DMA ----------------
// Workgroup tile: 64(M) x 256(N); 8 waves as 2x4; wave tile 32x64.
#define GEMM_Q 0
#define GEMM_K 1
#define GEMM_V 2
#define GEMM_O 3

#define KSTEP   32
#define KSTEPS  (D_MODEL / KSTEP)      // 64
#define AT_ELEM (64 * KSTEP)           // A tile elems (4 KiB bf16)
#define BT_ELEM (256 * KSTEP)          // B tile elems (16 KiB bf16)

// One pipeline stage: every thread issues 5 async b128 copies (1 A, 4 B).
__device__ __forceinline__ void gemm_stage_issue(
    const __bf16* __restrict__ A, const __bf16* __restrict__ Wt,
    unsigned int aLds, unsigned int bLds, int m0, int n0, int kk, int t)
{
  // A tile 64x32: thread t -> row t>>2, 16B chunk t&3
  const __bf16* ga = A + (size_t)(m0 + (t >> 2)) * D_MODEL + kk + (t & 3) * 8;
  async_b128(aLds + (unsigned int)((((t >> 2) * KSTEP) + (t & 3) * 8) * 2), ga);
  // B tile 256x32 (N-major): thread t -> row n0+t, 4 x 16B chunks
  const __bf16* gb = Wt + (size_t)(n0 + t) * D_MODEL + kk;
#pragma unroll
  for (int c = 0; c < 4; ++c)
    async_b128(bLds + (unsigned int)((t * KSTEP + c * 8) * 2), gb + c * 8);
}

__global__ __launch_bounds__(256) void gemm_wmma_kernel(
    const __bf16* __restrict__ A,    // [M_TOK][D_MODEL] bf16
    const __bf16* __restrict__ Wt,   // [D_MODEL n][D_MODEL k] bf16 (W transposed)
    const float*  __restrict__ bias, // [D_MODEL]
    __bf16* __restrict__ out_bf,     // Q/K/V modes
    float*  __restrict__ out_f,      // O mode
    int mode, float qscale)
{
  __shared__ __align__(16) __bf16 Abuf[2][AT_ELEM];   //  8 KiB
  __shared__ __align__(16) __bf16 Bbuf[2][BT_ELEM];   // 32 KiB
  const int t  = threadIdx.x;
  const int bm = blockIdx.x >> 3;                  // 64 M blocks
  const int bn = blockIdx.x & 7;                   // 8 N blocks
  const int m0 = bm * 64, n0 = bn * 256;
  const int wave   = t >> 5;
  const int wave_m = wave >> 2;                    // 0..1
  const int wave_n = wave & 3;                     // 0..3

  const unsigned int aLds[2] = { (unsigned int)(uintptr_t)&Abuf[0][0],
                                 (unsigned int)(uintptr_t)&Abuf[1][0] };
  const unsigned int bLds[2] = { (unsigned int)(uintptr_t)&Bbuf[0][0],
                                 (unsigned int)(uintptr_t)&Bbuf[1][0] };

  const v8f zero = {0.f, 0.f, 0.f, 0.f, 0.f, 0.f, 0.f, 0.f};
  v8f acc[2][4];
#pragma unroll
  for (int i = 0; i < 2; ++i)
#pragma unroll
    for (int j = 0; j < 4; ++j) acc[i][j] = zero;

  // prologue: stage K-step 0
  gemm_stage_issue(A, Wt, aLds[0], bLds[0], m0, n0, 0, t);

  for (int s = 0; s < KSTEPS; ++s) {
    wait_async0();          // step-s DMA landed in LDS
    __syncthreads();        // visible to all waves; prior compute done
    if (s + 1 < KSTEPS)     // kick DMA for step s+1 into the other buffer
      gemm_stage_issue(A, Wt, aLds[(s + 1) & 1], bLds[(s + 1) & 1],
                       m0, n0, (s + 1) * KSTEP, t);
    // compute step s from LDS while step s+1 is in flight
    const __bf16* Ab = &Abuf[s & 1][0];
    const __bf16* Bb = &Bbuf[s & 1][0];
    v16bf a0 = load_afrag(Ab, KSTEP, wave_m * 32, 0);
    v16bf a1 = load_afrag(Ab, KSTEP, wave_m * 32 + 16, 0);
#pragma unroll
    for (int j = 0; j < 4; ++j) {
      v16bf b = load_bfrag(Bb, KSTEP, wave_n * 64 + j * 16, 0);
      acc[0][j] = wmma_bf16(a0, b, acc[0][j]);
      acc[1][j] = wmma_bf16(a1, b, acc[1][j]);
    }
  }

  // epilogue
  const int lane  = t & 31;
  const int ncol  = lane & 15;
  const int rbase = (lane >> 4) << 3;             // 0 or 8
#pragma unroll
  for (int i = 0; i < 2; ++i) {
#pragma unroll
    for (int j = 0; j < 4; ++j) {
      const int ng = n0 + wave_n * 64 + j * 16 + ncol;    // output column
      const float bval = bias[ng];
#pragma unroll
      for (int r = 0; r < 8; ++r) {
        const int tok = m0 + wave_m * 32 + i * 16 + rbase + r;
        float v = acc[i][j][r] + bval;
        if (mode == GEMM_O) {
          out_f[(size_t)tok * D_MODEL + ng] = v;
        } else {
          if (mode == GEMM_Q) v *= qscale;
          const int b_ = tok >> 11;               // / SEQ
          const int s  = tok & (SEQ - 1);
          const int hh = ng >> 7;                 // / HEAD_DIM
          const int d  = ng & (HEAD_DIM - 1);
          const int bh = (b_ << 4) + hh;
          if (mode == GEMM_V)                     // V stored transposed [bh][d][s]
            out_bf[((size_t)bh * HEAD_DIM + d) * SEQ + s] = (__bf16)v;
          else                                    // Q/K stored [bh][s][d]
            out_bf[((size_t)bh * SEQ + s) * HEAD_DIM + d] = (__bf16)v;
        }
      }
    }
  }
}

// ---------------- 4) XPOS rotary (in-place on bf16 head-layout buffer) ----------------
__global__ __launch_bounds__(256) void xpos_kernel(__bf16* __restrict__ x, int downscale)
{
  const int idx = blockIdx.x * 256 + threadIdx.x;  // one (bh, s, pair)
  const int i  = idx & 63;                         // pair index, half = 64
  const int s  = (idx >> 6) & (SEQ - 1);
  const int bh = idx >> 17;                        // / (64*2048)
  if (bh >= BATCH * N_HEAD) return;
  const size_t base = ((size_t)bh * SEQ + s) * HEAD_DIM + 2 * i;
  const float x1 = (float)x[base], x2 = (float)x[base + 1];
  const float inv_freq = __powf(10000.f, -(float)i * (1.0f / 64.0f));
  const float ang = (float)s * inv_freq;
  float sn, cs;
  __sincosf(ang, &sn, &cs);
  const float sv = ((float)(2 * i) + 0.4f * HEAD_DIM) / (1.4f * HEAD_DIM);
  float p = ((float)s - (float)(SEQ / 2)) * (1.0f / 512.0f);
  if (downscale) p = -p;
  const float sc = __powf(sv, p);
  cs *= sc; sn *= sc;
  x[base]     = (__bf16)(x1 * cs - x2 * sn);
  x[base + 1] = (__bf16)(x2 * cs + x1 * sn);
}

// ---------------- 5) flash attention: 1 wave per 16-query tile ----------------
__global__ __launch_bounds__(32) void flash_attn_kernel(
    const __bf16* __restrict__ Q,   // [bh][s][128]  (already scaled)
    const __bf16* __restrict__ K,   // [bh][s][128]
    const __bf16* __restrict__ Vt,  // [bh][128][s]
    __bf16* __restrict__ attn)      // [b][s][h*128+d]
{
  __shared__ __align__(16) __bf16 ptile[16 * 32];
  const int lane  = threadIdx.x;
  const int nqt   = SEQ / 16;
  const int qtile = blockIdx.x % nqt;
  const int bh    = blockIdx.x / nqt;
  const int q0    = qtile * 16;
  const __bf16* Qh = Q  + (size_t)bh * SEQ * HEAD_DIM;
  const __bf16* Kh = K  + (size_t)bh * SEQ * HEAD_DIM;
  const __bf16* Vh = Vt + (size_t)bh * HEAD_DIM * SEQ;

  v16bf qa[4];
#pragma unroll
  for (int dc = 0; dc < 4; ++dc) qa[dc] = load_afrag(Qh, HEAD_DIM, q0, dc * 32);

  const v8f zero = {0.f, 0.f, 0.f, 0.f, 0.f, 0.f, 0.f, 0.f};
  v8f oacc[8];
  float mrun[8], lrun[8];
#pragma unroll
  for (int dt = 0; dt < 8; ++dt) oacc[dt] = zero;
#pragma unroll
  for (int r = 0; r < 8; ++r) { mrun[r] = -INFINITY; lrun[r] = 0.f; }

  const int ncol  = lane & 15;
  const int rbase = (lane >> 4) << 3;
  const int kbmax = (q0 + 15) >> 5;               // causal: 32-key blocks

  for (int kb = 0; kb <= kbmax; ++kb) {
    const int k0 = kb << 5;
    // warm WGP cache for the next key block (global_prefetch_b8)
    if (kb < kbmax) {
      __builtin_prefetch(Kh + (size_t)(k0 + 32 + lane) * HEAD_DIM, 0, 1);
      __builtin_prefetch(Vh + (size_t)(lane * 4) * SEQ + k0 + 32, 0, 1);
    }
    v8f sc0 = zero, sc1 = zero;
#pragma unroll
    for (int dc = 0; dc < 4; ++dc) {              // 8x wmma: S = Q K^T
      v16bf b0 = load_bfrag(Kh, HEAD_DIM, k0,      dc * 32);
      v16bf b1 = load_bfrag(Kh, HEAD_DIM, k0 + 16, dc * 32);
      sc0 = wmma_bf16(qa[dc], b0, sc0);
      sc1 = wmma_bf16(qa[dc], b1, sc1);
    }
    // online softmax on C-fragment layout (row r+rbase across 16-lane group)
#pragma unroll
    for (int r = 0; r < 8; ++r) {
      const int qrow = q0 + rbase + r;
      float s0 = (k0 + ncol      > qrow) ? -INFINITY : sc0[r];
      float s1 = (k0 + 16 + ncol > qrow) ? -INFINITY : sc1[r];
      float rowmax = fmaxf(s0, s1);
#pragma unroll
      for (int off = 8; off > 0; off >>= 1)
        rowmax = fmaxf(rowmax, __shfl_xor(rowmax, off, 32));
      const float mnew = fmaxf(mrun[r], rowmax);
      const float p0 = __expf(s0 - mnew);
      const float p1 = __expf(s1 - mnew);
      float psum = p0 + p1;
#pragma unroll
      for (int off = 8; off > 0; off >>= 1) psum += __shfl_xor(psum, off, 32);
      const float corr = __expf(mrun[r] - mnew);
      lrun[r] = lrun[r] * corr + psum;
      mrun[r] = mnew;
#pragma unroll
      for (int dt = 0; dt < 8; ++dt) oacc[dt][r] = oacc[dt][r] * corr;
      ptile[(rbase + r) * 32 + ncol]      = (__bf16)p0;
      ptile[(rbase + r) * 32 + 16 + ncol] = (__bf16)p1;
    }
    __syncthreads();   // single-wave WG: nop barrier, forces DScnt waits
    v16bf pa = load_afrag(&ptile[0], 32, 0, 0);   // C-layout -> A-layout via LDS
#pragma unroll
    for (int dt = 0; dt < 8; ++dt) {              // 8x wmma: O += P V
      v16bf bv_ = load_bfrag(Vh, SEQ, dt * 16, k0);
      oacc[dt] = wmma_bf16(pa, bv_, oacc[dt]);
    }
    __syncthreads();
  }

  const int b = bh >> 4, hh = bh & 15;
#pragma unroll
  for (int r = 0; r < 8; ++r) {
    const float inv = 1.0f / lrun[r];
    const int s = q0 + rbase + r;
    const size_t rowbase = ((size_t)(b * SEQ + s)) * D_MODEL + hh * HEAD_DIM;
#pragma unroll
    for (int dt = 0; dt < 8; ++dt)
      attn[rowbase + dt * 16 + ncol] = (__bf16)(oacc[dt][r] * inv);
  }
}

// ---------------- launch ----------------
extern "C" void kernel_launch(void* const* d_in, const int* in_sizes, int n_in,
                              void* d_out, int out_size, void* d_ws, size_t ws_size,
                              hipStream_t stream)
{
  (void)in_sizes; (void)n_in; (void)out_size; (void)ws_size;
  const float* h  = (const float*)d_in[0];
  const float* r  = (const float*)d_in[1];
  // d_in[2] = mask (all-true in this problem) -> no-op
  const float* nw = (const float*)d_in[3];
  const float* Wq = (const float*)d_in[4];  const float* bq = (const float*)d_in[5];
  const float* Wk = (const float*)d_in[6];  const float* bk = (const float*)d_in[7];
  const float* Wv = (const float*)d_in[8];  const float* bv = (const float*)d_in[9];
  const float* Wo = (const float*)d_in[10]; const float* bo = (const float*)d_in[11];

  float* out   = (float*)d_out;
  float* resid = out + (size_t)M_TOK * D_MODEL;

  char* ws = (char*)d_ws;
  const size_t SZ_X   = (size_t)M_TOK * D_MODEL * sizeof(__bf16);    // 16 MiB
  const size_t SZ_W   = (size_t)D_MODEL * D_MODEL * sizeof(__bf16);  //  8 MiB
  const size_t SZ_HQK = (size_t)BATCH * N_HEAD * SEQ * HEAD_DIM * sizeof(__bf16); // 16 MiB
  size_t off = 0;
  __bf16* xb    = (__bf16*)(ws + off); off += SZ_X;
  __bf16* wtq   = (__bf16*)(ws + off); off += SZ_W;
  __bf16* wtk   = (__bf16*)(ws + off); off += SZ_W;
  __bf16* wtv   = (__bf16*)(ws + off); off += SZ_W;
  __bf16* wto   = (__bf16*)(ws + off); off += SZ_W;
  __bf16* qb    = (__bf16*)(ws + off); off += SZ_HQK;
  __bf16* kbuf  = (__bf16*)(ws + off); off += SZ_HQK;
  __bf16* vt    = (__bf16*)(ws + off); off += SZ_HQK;
  __bf16* attnb = (__bf16*)(ws + off); off += SZ_X;                  // total ~112 MiB

  add_rmsnorm_kernel<<<M_TOK, 256, 0, stream>>>(h, r, nw, resid, xb);
  transpose_w_kernel<<<dim3(64, 64, 4), dim3(32, 8), 0, stream>>>(
      Wq, Wk, Wv, Wo, wtq, wtk, wtv, wto);

  const float qscale = 0.08838834764831845f;     // 128^-0.5
  // 512 blocks: 64 Mblocks(64) x 8 Nblocks(256); 8 waves / block
  gemm_wmma_kernel<<<512, 256, 0, stream>>>(xb, wtq, bq, qb,   nullptr, GEMM_Q, qscale);
  gemm_wmma_kernel<<<512, 256, 0, stream>>>(xb, wtk, bk, kbuf, nullptr, GEMM_K, 1.f);
  gemm_wmma_kernel<<<512, 256, 0, stream>>>(xb, wtv, bv, vt,   nullptr, GEMM_V, 1.f);

  xpos_kernel<<<16384, 256, 0, stream>>>(qb, 0);
  xpos_kernel<<<16384, 256, 0, stream>>>(kbuf, 1);

  flash_attn_kernel<<<BATCH * N_HEAD * (SEQ / 16), 32, 0, stream>>>(qb, kbuf, vt, attnb);

  gemm_wmma_kernel<<<512, 256, 0, stream>>>(attnb, wto, bo, nullptr, out, GEMM_O, 1.f);
}
